// SwinTransformerBlock_6004364280657
// MI455X (gfx1250) — matrix operand order repaired and checked by reading, not compiled, not run on hardware
//
#include <hip/hip_runtime.h>
#include <hip/hip_bf16.h>

// Swin block on gfx1250. All GEMMs via v_wmma_f32_16x16x32_f16 (wave32).
// B=32 H=W=56 C=192 WS=7 SHIFT=3 HEADS=6 HD=32 N=49 (padded to 64), NW=64.
// Activation tiles staged into LDS via TDM tensor_load_to_lds (attn) and
// global_load_async_to_lds_b128 (mlp) when the builtins are available.

typedef __attribute__((ext_vector_type(16))) _Float16 v16h;
typedef __attribute__((ext_vector_type(8)))  _Float16 v8h;
typedef __attribute__((ext_vector_type(8)))  float    v8f;
typedef unsigned int u32x4 __attribute__((ext_vector_type(4)));
typedef int          i32x4 __attribute__((ext_vector_type(4)));
typedef int          i32x8 __attribute__((ext_vector_type(8)));
typedef int          v4i_g __attribute__((vector_size(16)));   // builtin param type

#if __has_builtin(__builtin_amdgcn_tensor_load_to_lds)
#define HAS_TDM 1
#endif
#if __has_builtin(__builtin_amdgcn_global_load_async_to_lds_b128)
#define HAS_ASYNC_LDS 1
#endif

// Global (AS1) / LDS (AS3) v4i32 pointer casts for the async-LDS builtin.
#define G_V4(p) ((__attribute__((address_space(1))) v4i_g*)(uintptr_t)(p))
#define L_V4(p) ((__attribute__((address_space(3))) v4i_g*)(unsigned int)(uintptr_t)(p))

__device__ __forceinline__ void wait_asynccnt0() {
#if __has_builtin(__builtin_amdgcn_s_wait_asynccnt)
  __builtin_amdgcn_s_wait_asynccnt(0);
#else
  asm volatile("s_wait_asynccnt 0x0" ::: "memory");
#endif
}

// Stage `halfs` f16 elements global->LDS (block-wide, blockDim.x==256).
__device__ __forceinline__ void stage_tile(const _Float16* g, _Float16* l,
                                           int halfs) {
#ifdef HAS_ASYNC_LDS
  for (int i = threadIdx.x; i < (halfs >> 3); i += 256)
    __builtin_amdgcn_global_load_async_to_lds_b128(G_V4(g + i * 8),
                                                   L_V4(l + i * 8), 0, 0);
  wait_asynccnt0();
#else
  for (int i = threadIdx.x; i < (halfs >> 3); i += 256) {
    v8h tmp = *(const v8h*)(g + i * 8);
    *(v8h*)(l + i * 8) = tmp;
  }
#endif
  __syncthreads();
}

#ifdef HAS_TDM
// TDM: one DMA of a 2D f16 tile (rows x dim0, row stride = stride elems)
// from global into LDS.  Descriptor packed per CDNA5 ISA ch.8 (D# groups).
__device__ __forceinline__ void tdm_load_2d(const void* gaddr, void* lds,
                                            unsigned dim0, unsigned rows,
                                            unsigned stride) {
  unsigned long long ga = (unsigned long long)(uintptr_t)gaddr;
  u32x4 g0;
  g0[0] = 1u;                                   // count=1, user descriptor
  g0[1] = (unsigned)(uintptr_t)lds;             // lds_addr (byte offset)
  g0[2] = (unsigned)(ga & 0xffffffffu);         // global_addr[31:0]
  g0[3] = (unsigned)((ga >> 32) & 0x1ffffffu) | (2u << 30);  // addr hi | type=2
  i32x8 g1;
  g1[0] = (1 << 16);                            // data_size=1 (2B), mask=0
  g1[1] = (int)(dim0 << 16);                    // tensor_dim0[15:0]
  g1[2] = (int)(rows << 16);                    // dim0 hi=0 | tensor_dim1[15:0]
  g1[3] = (int)(dim0 << 16);                    // dim1 hi=0 | tile_dim0
  g1[4] = (int)rows;                            // tile_dim1 | tile_dim2=0
  g1[5] = (int)stride;                          // tensor_dim0_stride[31:0]
  g1[6] = 0;                                    // stride hi | dim1_stride lo
  g1[7] = 0;
  i32x4 g2 = {};
  i32x4 g3 = {};
#if defined(__clang_major__) && __clang_major__ >= 23
  i32x8 g4 = {};
  __builtin_amdgcn_tensor_load_to_lds(g0, g1, g2, g3, g4, 0);
#else
  __builtin_amdgcn_tensor_load_to_lds(g0, g1, g2, g3, 0);
#endif
}
#endif

// ---- WMMA fragment helpers (wave32; contiguous k order, not the ISA table order) ----------
// A (16x32 f16): lanes 0-15 row M=lane hold K=0..15;
// lanes 16-31 hold K=16..31.  -> one contiguous 16-half run (matches ldB).
__device__ __forceinline__ v16h ldA(const _Float16* p, int stride) {
  int lane = threadIdx.x & 31;
  const _Float16* q = p + (lane & 15) * stride + ((lane >> 4) << 4);
  v8h lo = *(const v8h*)q;
  v8h hi = *(const v8h*)(q + 8);
  v16h a;
#pragma unroll
  for (int i = 0; i < 8; ++i) { a[i] = lo[i]; a[i + 8] = hi[i]; }
  return a;
}
// B (32x16 f16) stored [n][k] row-major: lane holds column n=lane&15,
// contiguous K run of 16 starting at (lane>>4)*16.
__device__ __forceinline__ v16h ldB(const _Float16* p, int stride) {
  int lane = threadIdx.x & 31;
  return *(const v16h*)(p + (lane & 15) * stride + ((lane >> 4) << 4));
}
__device__ __forceinline__ v8f wmma16(v16h a, v16h b, v8f c) {
  return __builtin_amdgcn_wmma_f32_16x16x32_f16(false, a, false, b,
                                                (short)0, c, false, false);
}

// ---- K0: fp32 -> f16 weight conversion ------------------------------------
__global__ void k_cvt(const float* __restrict__ in, _Float16* __restrict__ out,
                      int n) {
  int i = blockIdx.x * blockDim.x + threadIdx.x;
  if (i < n) out[i] = (_Float16)in[i];
}

// ---- K1: LN1 + roll(-3,-3) + window partition -> xw f16 [2048][64][192] ----
__global__ __launch_bounds__(256, 1)
void k_ln1_window(const float* __restrict__ x, const float* __restrict__ g,
                  const float* __restrict__ bta, _Float16* __restrict__ xw) {
  int w = blockIdx.x;                 // window 0..2047
  int lane = threadIdx.x & 31, warp = threadIdx.x >> 5;
  int b = w >> 6, nw = w & 63, wh = nw >> 3, ww = nw & 7;
  for (int t = warp; t < 64; t += 8) {
    _Float16* dst = xw + (size_t)w * 12288 + t * 192;
    if (t < 49) {
      int r = t / 7, c = t % 7;
      int hi = (wh * 7 + r + 3) % 56;   // rolled[p] = x[(p+SHIFT)%56]
      int wi = (ww * 7 + c + 3) % 56;
      const float* src = x + ((size_t)b * 3136 + hi * 56 + wi) * 192;
      float v[6], s = 0.f;
#pragma unroll
      for (int j = 0; j < 6; ++j) { v[j] = src[lane + 32 * j]; s += v[j]; }
      for (int m = 16; m; m >>= 1) s += __shfl_xor(s, m, 32);
      float mu = s * (1.f / 192.f), vs = 0.f;
#pragma unroll
      for (int j = 0; j < 6; ++j) { float d = v[j] - mu; vs += d * d; }
      for (int m = 16; m; m >>= 1) vs += __shfl_xor(vs, m, 32);
      float rinv = rsqrtf(vs * (1.f / 192.f) + 1e-5f);
#pragma unroll
      for (int j = 0; j < 6; ++j) {
        int ch = lane + 32 * j;
        dst[ch] = (_Float16)((v[j] - mu) * rinv * g[ch] + bta[ch]);
      }
    } else {
#pragma unroll
      for (int j = 0; j < 6; ++j) dst[lane + 32 * j] = (_Float16)0.f;
    }
  }
}

// ---- K2: fused window attention -------------------------------------------
__global__ __launch_bounds__(256, 1)
void k_attn(const _Float16* __restrict__ xw, const _Float16* __restrict__ qkvw,
            const float* __restrict__ qkvb, const _Float16* __restrict__ projw,
            const float* __restrict__ projb, const float* __restrict__ rpb,
            const float* __restrict__ mask, const float* __restrict__ xin,
            float* __restrict__ xres) {
  extern __shared__ char smem[];
  _Float16* a_s = (_Float16*)smem;      // [64][192] staged activations
  _Float16* q_s = a_s + 12288;          // [6][64][32]
  _Float16* k_s = q_s + 12288;          // [6][64][32]
  _Float16* v_s = k_s + 12288;          // [6][32][64]  (d-major!)
  _Float16* o_s = v_s + 12288;          // [64][192]
  _Float16* p_s = o_s + 12288;          // [64][64]
  float*    s_s = (float*)(p_s + 4096); // [64][64]
  int w = blockIdx.x;
  int lane = threadIdx.x & 31, wv = threadIdx.x >> 5;
  int b = w >> 6, nw = w & 63, wh = nw >> 3, ww = nw & 7;
  const _Float16* A0 = xw + (size_t)w * 12288;

  // Stage the 64x192 f16 window tile into LDS.
#ifdef HAS_TDM
  if (wv == 0) {
    tdm_load_2d(A0, a_s, 192, 64, 192);
    __builtin_amdgcn_s_wait_tensorcnt(0);
  }
  __syncthreads();
#else
  stage_tile(A0, a_s, 12288);
#endif

  // Phase 1: QKV = xw @ qkv_w^T   (64x192)x(192x576), route into LDS
  for (int t = wv; t < 144; t += 8) {
    int mt = t / 36, nt = t % 36;
    v8f acc = {};
#pragma unroll
    for (int kf = 0; kf < 6; ++kf)
      acc = wmma16(ldA(a_s + mt * 16 * 192 + kf * 32, 192),
                   ldB(qkvw + nt * 16 * 192 + kf * 32, 192), acc);
    int n = lane & 15, mb = (lane >> 4) << 3;
    int ngl = nt * 16 + n;
    int which = ngl / 192, hh = (ngl % 192) >> 5, d = ngl & 31;
    float bias = qkvb[ngl];
#pragma unroll
    for (int r = 0; r < 8; ++r) {
      int m = mt * 16 + mb + r;
      float val = acc[r] + bias;
      if (which == 0)       q_s[hh * 2048 + m * 32 + d] = (_Float16)(val * 0.17677669529f);
      else if (which == 1)  k_s[hh * 2048 + m * 32 + d] = (_Float16)val;
      else                  v_s[hh * 2048 + d * 64 + m] = (_Float16)val;
    }
  }
  __syncthreads();

  // Phase 2: per-head attention
  for (int h = 0; h < 6; ++h) {
    // S = Q K^T  (64x32)x(32x64)
    for (int t = wv; t < 16; t += 8) {
      int mt = t >> 2, nt = t & 3;
      v8f acc = {};
      acc = wmma16(ldA(q_s + h * 2048 + mt * 16 * 32, 32),
                   ldB(k_s + h * 2048 + nt * 16 * 32, 32), acc);
      int n = lane & 15, mb = (lane >> 4) << 3;
#pragma unroll
      for (int r = 0; r < 8; ++r)
        s_s[(mt * 16 + mb + r) * 64 + nt * 16 + n] = acc[r];
    }
    __syncthreads();
    // rel-pos bias + shift mask + padding mask
    for (int idx = threadIdx.x; idx < 4096; idx += 256) {
      int i = idx >> 6, j = idx & 63;
      float v = s_s[idx];
      if (j >= 49) v = -1e30f;
      else if (i < 49) {
        int r1 = i / 7, c1 = i % 7, r2 = j / 7, c2 = j % 7;
        int ridx = (r1 - r2 + 6) * 13 + (c1 - c2 + 6);
        v += rpb[ridx * 6 + h] + mask[(nw * 49 + i) * 49 + j];
      }
      s_s[idx] = v;
    }
    __syncthreads();
    // softmax per row -> P (f16)
    if (threadIdx.x < 64) {
      float* row = s_s + threadIdx.x * 64;
      float mx = -1e30f;
      for (int j = 0; j < 64; ++j) mx = fmaxf(mx, row[j]);
      float sm = 0.f;
      for (int j = 0; j < 64; ++j) sm += __expf(row[j] - mx);
      float inv = 1.f / sm;
      _Float16* prow = p_s + threadIdx.x * 64;
      for (int j = 0; j < 64; ++j)
        prow[j] = (_Float16)(__expf(row[j] - mx) * inv);
    }
    __syncthreads();
    // O_h = P V   (64x64)x(64x32); V stored [d][j] so ldB is contiguous
    for (int t = wv; t < 8; t += 8) {
      int mt = t >> 1, nt = t & 1;
      v8f acc = {};
#pragma unroll
      for (int kf = 0; kf < 2; ++kf)
        acc = wmma16(ldA(p_s + mt * 16 * 64 + kf * 32, 64),
                     ldB(v_s + h * 2048 + nt * 16 * 64 + kf * 32, 64), acc);
      int n = lane & 15, mb = (lane >> 4) << 3;
#pragma unroll
      for (int r = 0; r < 8; ++r)
        o_s[(mt * 16 + mb + r) * 192 + h * 32 + nt * 16 + n] = (_Float16)acc[r];
    }
    __syncthreads();
  }

  // Phase 3: proj + window-reverse + roll(+3,+3) + residual
  for (int t = wv; t < 48; t += 8) {
    int mt = t / 12, nt = t % 12;
    v8f acc = {};
#pragma unroll
    for (int kf = 0; kf < 6; ++kf)
      acc = wmma16(ldA(o_s + mt * 16 * 192 + kf * 32, 192),
                   ldB(projw + nt * 16 * 192 + kf * 32, 192), acc);
    int n = lane & 15, mb = (lane >> 4) << 3;
    int ngl = nt * 16 + n;
    float bias = projb[ngl];
#pragma unroll
    for (int r = 0; r < 8; ++r) {
      int tok = mt * 16 + mb + r;
      if (tok < 49) {
        int rr = tok / 7, cc = tok % 7;
        int hi = (wh * 7 + rr + 3) % 56;  // same map as forward gather
        int wi = (ww * 7 + cc + 3) % 56;
        size_t gi = ((size_t)b * 3136 + hi * 56 + wi) * 192 + ngl;
        xres[gi] = xin[gi] + acc[r] + bias;
      }
    }
  }
}

// ---- K3: LN2 -> f16 --------------------------------------------------------
__global__ __launch_bounds__(256, 1)
void k_ln2(const float* __restrict__ xres, const float* __restrict__ g,
           const float* __restrict__ bta, _Float16* __restrict__ h2) {
  int tok = blockIdx.x * 8 + (threadIdx.x >> 5);
  int lane = threadIdx.x & 31;
  if (tok >= 100352) return;
  const float* src = xres + (size_t)tok * 192;
  float v[6], s = 0.f;
#pragma unroll
  for (int j = 0; j < 6; ++j) { v[j] = src[lane + 32 * j]; s += v[j]; }
  for (int m = 16; m; m >>= 1) s += __shfl_xor(s, m, 32);
  float mu = s * (1.f / 192.f), vs = 0.f;
#pragma unroll
  for (int j = 0; j < 6; ++j) { float d = v[j] - mu; vs += d * d; }
  for (int m = 16; m; m >>= 1) vs += __shfl_xor(vs, m, 32);
  float rinv = rsqrtf(vs * (1.f / 192.f) + 1e-5f);
  _Float16* dst = h2 + (size_t)tok * 192;
#pragma unroll
  for (int j = 0; j < 6; ++j) {
    int ch = lane + 32 * j;
    dst[ch] = (_Float16)((v[j] - mu) * rinv * g[ch] + bta[ch]);
  }
}

// ---- K4: fused MLP (fc1 + GELU in LDS, fc2 + residual) --------------------
__global__ __launch_bounds__(256, 1)
void k_mlp(const _Float16* __restrict__ h2, const _Float16* __restrict__ fc1w,
           const float* __restrict__ fc1b, const _Float16* __restrict__ fc2w,
           const float* __restrict__ fc2b, const float* __restrict__ xres,
           float* __restrict__ out) {
  extern __shared__ char smem[];
  _Float16* a_s = (_Float16*)smem;  // [64][192] staged activations
  _Float16* y1 = a_s + 12288;       // [64][768]
  size_t tok0 = (size_t)blockIdx.x * 64;
  int lane = threadIdx.x & 31, wv = threadIdx.x >> 5;

  stage_tile(h2 + tok0 * 192, a_s, 12288);  // async -> LDS

  for (int t = wv; t < 192; t += 8) {       // 64x768 output tiles
    int mt = t / 48, nt = t % 48;
    v8f acc = {};
#pragma unroll
    for (int kf = 0; kf < 6; ++kf)
      acc = wmma16(ldA(a_s + mt * 16 * 192 + kf * 32, 192),
                   ldB(fc1w + nt * 16 * 192 + kf * 32, 192), acc);
    int n = lane & 15, mb = (lane >> 4) << 3;
    float bias = fc1b[nt * 16 + n];
#pragma unroll
    for (int r = 0; r < 8; ++r) {
      float v = acc[r] + bias;
      v = 0.5f * v * (1.f + erff(v * 0.70710678118f));   // exact GELU
      y1[(mt * 16 + mb + r) * 768 + nt * 16 + n] = (_Float16)v;
    }
  }
  __syncthreads();
  for (int t = wv; t < 48; t += 8) {        // 64x192 output tiles, K=768
    int mt = t / 12, nt = t % 12;
    v8f acc = {};
#pragma unroll
    for (int kf = 0; kf < 24; ++kf)
      acc = wmma16(ldA(y1 + mt * 16 * 768 + kf * 32, 768),
                   ldB(fc2w + nt * 16 * 768 + kf * 32, 768), acc);
    int n = lane & 15, mb = (lane >> 4) << 3;
    float bias = fc2b[nt * 16 + n];
#pragma unroll
    for (int r = 0; r < 8; ++r) {
      size_t gi = (tok0 + mt * 16 + mb + r) * 192 + nt * 16 + n;
      out[gi] = xres[gi] + acc[r] + bias;
    }
  }
}

// ---- host launch -----------------------------------------------------------
extern "C" void kernel_launch(void* const* d_in, const int* in_sizes, int n_in,
                              void* d_out, int out_size, void* d_ws,
                              size_t ws_size, hipStream_t stream) {
  (void)in_sizes; (void)n_in; (void)out_size; (void)ws_size;
  const float* x       = (const float*)d_in[0];
  const float* mask    = (const float*)d_in[1];
  const float* n1g     = (const float*)d_in[2];
  const float* n1b     = (const float*)d_in[3];
  const float* qkv_w   = (const float*)d_in[4];
  const float* qkv_b   = (const float*)d_in[5];
  const float* proj_w  = (const float*)d_in[6];
  const float* proj_b  = (const float*)d_in[7];
  const float* rpb     = (const float*)d_in[8];
  const float* n2g     = (const float*)d_in[9];
  const float* n2b     = (const float*)d_in[10];
  const float* fc1_w   = (const float*)d_in[11];
  const float* fc1_b   = (const float*)d_in[12];
  const float* fc2_w   = (const float*)d_in[13];
  const float* fc2_b   = (const float*)d_in[14];
  float* out = (float*)d_out;

  uint8_t* ws = (uint8_t*)d_ws;
  _Float16* xw    = (_Float16*)(ws);                       // 50,331,648 B
  float*    xres  = (float*)   (ws + 50331648);            // 77,070,336 B
  _Float16* h2    = (_Float16*)(ws + 127401984);           // 38,535,168 B
  _Float16* qkvwh = (_Float16*)(ws + 165937152);           //    221,184 B
  _Float16* projwh= (_Float16*)(ws + 166158336);           //     73,728 B
  _Float16* fc1wh = (_Float16*)(ws + 166232064);           //    294,912 B
  _Float16* fc2wh = (_Float16*)(ws + 166526976);           //    294,912 B

  k_cvt<<<(110592 + 255) / 256, 256, 0, stream>>>(qkv_w, qkvwh, 110592);
  k_cvt<<<(36864 + 255) / 256, 256, 0, stream>>>(proj_w, projwh, 36864);
  k_cvt<<<(147456 + 255) / 256, 256, 0, stream>>>(fc1_w, fc1wh, 147456);
  k_cvt<<<(147456 + 255) / 256, 256, 0, stream>>>(fc2_w, fc2wh, 147456);

  k_ln1_window<<<2048, 256, 0, stream>>>(x, n1g, n1b, xw);
  k_attn<<<2048, 256, 147456, stream>>>(xw, qkvwh, qkv_b, projwh, proj_b,
                                        rpb, mask, x, xres);
  k_ln2<<<12544, 256, 0, stream>>>(xres, n2g, n2b, h2);
  k_mlp<<<1568, 256, 122880, stream>>>(h2, fc1wh, fc1_b, fc2wh, fc2_b,
                                       xres, out);
}
